// EdgewiseMSA_75093208203520
// MI455X (gfx1250) — compile-verified, matrix operand order checked by reading, not run on hardware
//
#include <hip/hip_runtime.h>
#include <hip/hip_bf16.h>
#include <math.h>

// ---------------------------------------------------------------------------
// EdgewiseMSA for MI455X (gfx1250). All GEMMs via v_wmma_f32_16x16x32_bf16,
// TN form (both operands K-contiguous -> fragment reads are ds_load_b128),
// global->LDS staging via gfx1250 async-to-LDS engine (ASYNCcnt), double-
// buffered LDS. B=2, N=768, D=512, H=8, DK=64. ~45 GFLOP, compute-bound.
// ---------------------------------------------------------------------------

typedef __attribute__((ext_vector_type(16))) __bf16 bf16x16;
typedef __attribute__((ext_vector_type(8)))  float  f32x8;

#define HH   8
#define NT   768
#define DM   512
#define DKH  64
#define BB   2
#define TD   1536          // 3*DM
#define BNROWS 1536        // B*N

// ---------------------------------------------------------------------------
// f32 -> bf16 convert (plain)
// ---------------------------------------------------------------------------
__global__ __launch_bounds__(256) void cvt_bf16_kernel(const float* __restrict__ in,
                                                       __bf16* __restrict__ out, long n) {
    long i = (long)blockIdx.x * 256 + threadIdx.x;
    if (i < n) out[i] = (__bf16)in[i];
}

// ---------------------------------------------------------------------------
// f32 (R x C, row-major) -> bf16 transposed (C x R). For weight matrices.
// ---------------------------------------------------------------------------
__global__ __launch_bounds__(256)
void cvtT_bf16_kernel(const float* __restrict__ in, __bf16* __restrict__ out,
                      int R, int C) {
    long i = (long)blockIdx.x * 256 + threadIdx.x;
    if (i < (long)R * C) {
        int k = (int)(i / C), n = (int)(i - (long)k * C);
        out[(long)n * R + k] = (__bf16)in[i];
    }
}

// ---------------------------------------------------------------------------
// bf16 tiled transpose (64x64 tiles through LDS), batched (z = b*H + h).
// ---------------------------------------------------------------------------
__global__ __launch_bounds__(256)
void transpose_bf16_kernel(const __bf16* __restrict__ in, __bf16* __restrict__ out,
                           int ldi, int ldo,
                           long sIb, long sIh, long sOb, long sOh) {
    __shared__ __bf16 tile[64 * 72];
    const int z = blockIdx.z;
    const __bf16* ip = in  + (long)(z / HH) * sIb + (long)(z % HH) * sIh;
    __bf16*       op = out + (long)(z / HH) * sOb + (long)(z % HH) * sOh;
    const int r0 = blockIdx.x * 64, c0 = blockIdx.y * 64;
    const int tid = threadIdx.x;

    #pragma unroll
    for (int v = tid; v < 512; v += 256) {
        int r = v >> 3, c8 = (v & 7) << 3;
        *(uint4*)&tile[r * 72 + c8] = *(const uint4*)&ip[(long)(r0 + r) * ldi + c0 + c8];
    }
    __syncthreads();
    #pragma unroll
    for (int v = tid; v < 512; v += 256) {
        int c = v >> 3, r8 = (v & 7) << 3;
        union { uint4 q; __bf16 e[8]; } u;
        #pragma unroll
        for (int i = 0; i < 8; ++i) u.e[i] = tile[(r8 + i) * 72 + c];
        *(uint4*)&op[(long)(c0 + c) * ldo + r0 + r8] = u.q;
    }
}

// ---------------------------------------------------------------------------
// TN bf16 WMMA GEMM with async-to-LDS double-buffered staging.
//   C[z] = alpha * A[z](MxK, row-major) @ B[z]^T,  B stored N x K row-major.
//   WN = wave N-tiles (2 -> BN=64, 4 -> BN=128). Block: 128 x (WN*32),
//   8 waves in a 4x2 grid, each wave 32 x (WN*16)... wave tile = 32 x WN*16*? 
//   Wave covers 32 rows (2 M-tiles) x WN*16 cols (WN N-tiles) = 2*WN WMMAs.
//   TRANS_C: store C transposed.
// Requires M%128==0, N%(WN*32)==0, K%32==0 (true for all uses here).
// ---------------------------------------------------------------------------
template<int WN, bool OUT_BF16, bool TRANS_C>
__global__ __launch_bounds__(256)
void gemm_async_wmma(const __bf16* __restrict__ A, const __bf16* __restrict__ B,
                     void* __restrict__ Cmat, int K,
                     int lda, int ldb, int ldc,
                     long sAb, long sAh, long sBb, long sBh, long sCb, long sCh,
                     float alpha)
{
    constexpr int BM = 128, BN = WN * 32, BK = 32, LD = BK + 8;
    constexpr int ASZ = BM * LD, BSZ = BN * LD, BUFSZ = ASZ + BSZ;
    constexpr int NBJ = BN / 64;                 // B staging chunks per thread
    constexpr int NASYNC = 2 + NBJ;              // async instrs / wave / chunk
    __shared__ __bf16 smem[2 * BUFSZ];           // single LDS symbol: offset 0

    const int z = blockIdx.z;
    const __bf16* Ab = A + (long)(z / HH) * sAb + (long)(z % HH) * sAh
                         + (long)blockIdx.x * BM * lda;
    const __bf16* Bb = B + (long)(z / HH) * sBb + (long)(z % HH) * sBh
                         + (long)blockIdx.y * BN * ldb;
    const long oC = (long)(z / HH) * sCb + (long)(z % HH) * sCh;

    const int tid = threadIdx.x, lane = tid & 31, wave = tid >> 5;
    const int wm = wave & 3, wn = wave >> 2;          // 4x2 wave grid
    const int kb = (lane < 16) ? 0 : 8, l15 = lane & 15;

    // staging coords: rows tid>>2 (+64), 16B column chunk (tid&3)*8
    const int sr = tid >> 2, sc = (tid & 3) << 3;

    // issue one K-chunk of async global->LDS copies (16B per lane per instr)
    auto issue = [&](int k0, int buf) {
        #pragma unroll
        for (int j = 0; j < 2; ++j) {
            unsigned long long g =
                (unsigned long long)&Ab[(long)(sr + 64 * j) * lda + k0 + sc];
            unsigned l = (unsigned)((buf * BUFSZ + (sr + 64 * j) * LD + sc) * 2);
            asm volatile("global_load_async_to_lds_b128 %0, %1, off"
                         :: "v"(l), "v"(g) : "memory");
        }
        #pragma unroll
        for (int j = 0; j < NBJ; ++j) {
            unsigned long long g =
                (unsigned long long)&Bb[(long)(sr + 64 * j) * ldb + k0 + sc];
            unsigned l = (unsigned)((buf * BUFSZ + ASZ + (sr + 64 * j) * LD + sc) * 2);
            asm volatile("global_load_async_to_lds_b128 %0, %1, off"
                         :: "v"(l), "v"(g) : "memory");
        }
    };

    f32x8 acc[2][WN] = {};
    issue(0, 0);
    int buf = 0;
    for (int k0 = 0; k0 < K; k0 += BK) {
        const bool more = (k0 + BK) < K;
        if (more) {
            issue(k0 + BK, buf ^ 1);               // prefetch next chunk
            asm volatile("s_wait_asynccnt %0" :: "n"(NASYNC) : "memory");
        } else {
            asm volatile("s_wait_asynccnt 0" ::: "memory");
        }
        __syncthreads();                            // all waves' copies visible

        const __bf16* As = smem + buf * BUFSZ;
        const __bf16* Bs = smem + buf * BUFSZ + ASZ;
        union { bf16x16 v; uint4 q[2]; } af[2], bfr[WN];
        #pragma unroll
        for (int r = 0; r < 2; ++r) {
            int m = (wm * 32 + r * 16 + l15) * LD + kb;
            af[r].q[0] = *(const uint4*)&As[m];
            af[r].q[1] = *(const uint4*)&As[m + 16];
        }
        #pragma unroll
        for (int c = 0; c < WN; ++c) {
            int n = (wn * (WN * 16) + c * 16 + l15) * LD + kb;
            bfr[c].q[0] = *(const uint4*)&Bs[n];
            bfr[c].q[1] = *(const uint4*)&Bs[n + 16];
        }
        #pragma unroll
        for (int r = 0; r < 2; ++r)
            #pragma unroll
            for (int c = 0; c < WN; ++c)
                acc[r][c] = __builtin_amdgcn_wmma_f32_16x16x32_bf16(
                    false, af[r].v, false, bfr[c].v, (short)0, acc[r][c],
                    false, false);
        __syncthreads();                            // protect buf before reuse
        buf ^= 1;
    }

    // store: C VGPR v -> M = v (+8 for hi lanes), N = lane&15
    const int m0 = blockIdx.x * BM, n0 = blockIdx.y * BN;
    #pragma unroll
    for (int r = 0; r < 2; ++r) {
        #pragma unroll
        for (int c = 0; c < WN; ++c) {
            union { f32x8 v; float e[8]; } u; u.v = acc[r][c];
            int mb = m0 + wm * 32 + r * 16 + ((lane < 16) ? 0 : 8);
            int nc = n0 + wn * (WN * 16) + c * 16 + l15;
            #pragma unroll
            for (int vi = 0; vi < 8; ++vi) {
                float val = u.e[vi] * alpha;
                long ci = TRANS_C ? oC + (long)nc * ldc + (mb + vi)
                                  : oC + (long)(mb + vi) * ldc + nc;
                if (OUT_BF16) ((__bf16*)Cmat)[ci] = (__bf16)val;
                else          ((float*)Cmat)[ci]  = val;
            }
        }
    }
}

// ---------------------------------------------------------------------------
// Row softmax (row length NT=768), f32 in -> bf16 out. 1 block / row.
// ---------------------------------------------------------------------------
__global__ __launch_bounds__(256)
void softmax_rows_kernel(const float* __restrict__ S, __bf16* __restrict__ Aout) {
    __shared__ float redm[8], reds[8];
    const long row = blockIdx.x;
    const float* s = S + row * NT;
    __bf16* a = Aout + row * NT;
    const int tid = threadIdx.x, lane = tid & 31, wave = tid >> 5;

    float v[3], mx = -3.4e38f;
    #pragma unroll
    for (int i = 0; i < 3; ++i) { v[i] = s[tid + i * 256]; mx = fmaxf(mx, v[i]); }
    #pragma unroll
    for (int off = 16; off; off >>= 1) mx = fmaxf(mx, __shfl_xor(mx, off, 32));
    if (lane == 0) redm[wave] = mx;
    __syncthreads();
    mx = redm[0];
    #pragma unroll
    for (int i = 1; i < 8; ++i) mx = fmaxf(mx, redm[i]);

    float sum = 0.f;
    #pragma unroll
    for (int i = 0; i < 3; ++i) { v[i] = __expf(v[i] - mx); sum += v[i]; }
    #pragma unroll
    for (int off = 16; off; off >>= 1) sum += __shfl_xor(sum, off, 32);
    if (lane == 0) reds[wave] = sum;
    __syncthreads();
    sum = 0.f;
    #pragma unroll
    for (int i = 0; i < 8; ++i) sum += reds[i];
    float inv = 1.f / sum;
    #pragma unroll
    for (int i = 0; i < 3; ++i) a[tid + i * 256] = (__bf16)(v[i] * inv);
}

// ---------------------------------------------------------------------------
// Per-edge gate MLP + Smix (in-place over Cr). idx = ((b*H+h)*N + n)*N + m
// ---------------------------------------------------------------------------
__global__ __launch_bounds__(256)
void gate_smix_kernel(const float* __restrict__ S1, const float* __restrict__ S2,
                      float* __restrict__ CrSmix, const float* __restrict__ Cl,
                      const float* __restrict__ w1, const float* __restrict__ b1,
                      const float* __restrict__ w2, const float* __restrict__ b2)
{
    __shared__ float c1w[96], c1b[16], c2w[64], c2b[4];
    const int t = threadIdx.x;
    if (t < 96) c1w[t] = w1[t];
    if (t < 16) c1b[t] = b1[t];
    if (t < 64) c2w[t] = w2[t];
    if (t < 4)  c2b[t] = b2[t];
    __syncthreads();

    const long idx = (long)blockIdx.x * 256 + t;
    const long bh  = idx / ((long)NT * NT);
    const int  rem = (int)(idx - bh * (long)NT * NT);
    const int  n   = rem / NT, m = rem - n * NT;
    const long tix = bh * (long)NT * NT + (long)m * NT + n;

    float s1 = S1[idx], s2 = S2[idx];
    float feat[6];
    feat[0] = s1;
    feat[1] = s2;
    feat[2] = S1[tix];
    feat[3] = S2[tix];
    feat[4] = __logf(CrSmix[idx] + 1e-6f);
    feat[5] = __logf(Cl[idx] + 1e-6f);

    float hid[16];
    #pragma unroll
    for (int o = 0; o < 16; ++o) {
        float a = c1b[o];
        #pragma unroll
        for (int c = 0; c < 6; ++c) a += c1w[o * 6 + c] * feat[c];
        float th = tanhf(0.7978845608f * (a + 0.044715f * a * a * a));
        hid[o] = 0.5f * a * (1.f + th);                 // tanh-gelu
    }
    float g[4];
    #pragma unroll
    for (int gi = 0; gi < 4; ++gi) {
        float a = c2b[gi];
        #pragma unroll
        for (int o = 0; o < 16; ++o) a += c2w[gi * 16 + o] * hid[o];
        g[gi] = 1.f / (1.f + __expf(-a));               // sigmoid
    }
    float mx  = fmaxf(s1, s2);
    float lae = mx + log1pf(__expf(-fabsf(s1 - s2)));   // logaddexp
    CrSmix[idx] = s1 + g[0] * s2 + g[1] * (lae - s1) - g[2] * (0.5f * s2)
                + g[3] * feat[4];
}

// ---------------------------------------------------------------------------
// y = y_base + sigmoid(chain_logit) * y_chain -> bf16
// ---------------------------------------------------------------------------
__global__ __launch_bounds__(256)
void combine_kernel(const float* __restrict__ yb, const float* __restrict__ yc,
                    const float* __restrict__ chain, __bf16* __restrict__ out, long n) {
    long i = (long)blockIdx.x * 256 + threadIdx.x;
    if (i < n) {
        float w = 1.f / (1.f + __expf(-chain[0]));
        out[i] = (__bf16)(yb[i] + w * yc[i]);
    }
}

// ---------------------------------------------------------------------------
extern "C" void kernel_launch(void* const* d_in, const int* in_sizes, int n_in,
                              void* d_out, int out_size, void* d_ws, size_t ws_size,
                              hipStream_t stream) {
    const float* x   = (const float*)d_in[0];
    const float* w1  = (const float*)d_in[1];
    const float* w2  = (const float*)d_in[2];
    const float* pw  = (const float*)d_in[3];
    const float* c1w = (const float*)d_in[4];
    const float* c1b = (const float*)d_in[5];
    const float* c2w = (const float*)d_in[6];
    const float* c2b = (const float*)d_in[7];
    const float* chl = (const float*)d_in[8];
    float* out = (float*)d_out;

    const long SZ_X   = (long)BNROWS * DM;        // 786432
    const long SZ_W   = (long)DM * TD;            // 786432
    const long SZ_PW  = (long)DM * DM;            // 262144
    const long SZ_QKV = (long)BNROWS * TD;        // 2359296
    const long SZ_S   = (long)BB * HH * NT * NT;  // 9437184
    const long SZ_T   = (long)BB * HH * NT * DKH; // 786432
    const long SZ_Y   = (long)BB * NT * DM;       // 786432

    char* p = (char*)d_ws;
    auto alloc = [&](size_t bytes) -> void* {
        void* r = (void*)p; p += (bytes + 255) & ~(size_t)255; return r;
    };
    __bf16* x_bf    = (__bf16*)alloc(SZ_X * 2);
    __bf16* w1T     = (__bf16*)alloc(SZ_W * 2);
    __bf16* w2T     = (__bf16*)alloc(SZ_W * 2);
    __bf16* pwT     = (__bf16*)alloc(SZ_PW * 2);
    __bf16* qkv1_bf = (__bf16*)alloc(SZ_QKV * 2);
    __bf16* qkv2_bf = (__bf16*)alloc(SZ_QKV * 2);
    __bf16* v1T     = (__bf16*)alloc(SZ_T * 2);
    __bf16* v2T     = (__bf16*)alloc(SZ_T * 2);
    float*  S1      = (float*) alloc(SZ_S * 4);
    float*  S2      = (float*) alloc(SZ_S * 4);
    __bf16* A1bf    = (__bf16*)alloc(SZ_S * 2);
    __bf16* A2bf    = (__bf16*)alloc(SZ_S * 2);
    __bf16* A1T     = (__bf16*)alloc(SZ_S * 2);
    __bf16* A2T     = (__bf16*)alloc(SZ_S * 2);
    float*  Cr      = (float*) alloc(SZ_S * 4);   // becomes Smix in-place
    float*  Cl      = (float*) alloc(SZ_S * 4);
    __bf16* Tt      = (__bf16*)alloc(SZ_T * 2);   // transport^T (B,H,DK,N)
    // recycled regions (dead by the time they are rewritten):
    __bf16* Abf     = A2T;              // final softmax out (A2T read is done)
    float*  ychain  = S1;               // S1 dead after gate kernel
    float*  ybase   = S2;               // S2 dead after gate kernel
    __bf16* y_bf    = (__bf16*)Cl;      // Cl dead after gate kernel

    const long sSb = (long)HH * NT * NT,  sSh = (long)NT * NT;   // (B,H,N,N)
    const long sQb = (long)NT * TD,       sQh = DKH;             // qkv views
    const long sTb = (long)HH * DKH * NT, sTh = (long)DKH * NT;  // (B,H,DK,N)
    const long sYb = (long)NT * DM,       sYh = DKH;             // (B,N,H,DK)
    const float scale = 0.125f;                                  // 1/sqrt(64)

    // ---- precision conversion / weight transposes (K-contiguous operands)
    cvt_bf16_kernel<<<dim3(3072), 256, 0, stream>>>(x, x_bf, SZ_X);
    cvtT_bf16_kernel<<<dim3(3072), 256, 0, stream>>>(w1, w1T, DM, TD);
    cvtT_bf16_kernel<<<dim3(3072), 256, 0, stream>>>(w2, w2T, DM, TD);
    cvtT_bf16_kernel<<<dim3(1024), 256, 0, stream>>>(pw, pwT, DM, DM);

    // ---- QKV projections: (1536x512) @ (512x1536) -> bf16
    gemm_async_wmma<4, true, false><<<dim3(12, 12, 1), 256, 0, stream>>>(
        x_bf, w1T, qkv1_bf, DM, DM, DM, TD, 0, 0, 0, 0, 0, 0, 1.f);
    gemm_async_wmma<4, true, false><<<dim3(12, 12, 1), 256, 0, stream>>>(
        x_bf, w2T, qkv2_bf, DM, DM, DM, TD, 0, 0, 0, 0, 0, 0, 1.f);

    // ---- S = scale * Q @ K^T (16 batches of 768x768x64; K rows are N x DK)
    gemm_async_wmma<4, false, false><<<dim3(6, 6, 16), 256, 0, stream>>>(
        qkv1_bf, qkv1_bf + DM, S1, DKH, TD, TD, NT,
        sQb, sQh, sQb, sQh, sSb, sSh, scale);
    gemm_async_wmma<4, false, false><<<dim3(6, 6, 16), 256, 0, stream>>>(
        qkv2_bf, qkv2_bf + DM, S2, DKH, TD, TD, NT,
        sQb, sQh, sQb, sQh, sSb, sSh, scale);

    // ---- A1, A2 = softmax -> bf16
    softmax_rows_kernel<<<dim3(BB * HH * NT), 256, 0, stream>>>(S1, A1bf);
    softmax_rows_kernel<<<dim3(BB * HH * NT), 256, 0, stream>>>(S2, A2bf);

    // ---- transposes for K-contiguous B operands
    transpose_bf16_kernel<<<dim3(12, 12, 16), 256, 0, stream>>>(
        A1bf, A1T, NT, NT, sSb, sSh, sSb, sSh);
    transpose_bf16_kernel<<<dim3(12, 12, 16), 256, 0, stream>>>(
        A2bf, A2T, NT, NT, sSb, sSh, sSb, sSh);
    transpose_bf16_kernel<<<dim3(12, 1, 16), 256, 0, stream>>>(
        qkv1_bf + 2 * DM, v1T, TD, NT, sQb, sQh, sTb, sTh);
    transpose_bf16_kernel<<<dim3(12, 1, 16), 256, 0, stream>>>(
        qkv2_bf + 2 * DM, v2T, TD, NT, sQb, sQh, sTb, sTh);

    // ---- Cr = A1@A2, Cl = A2@A1 (16 batches of 768^3 — dominant cost)
    gemm_async_wmma<4, false, false><<<dim3(6, 6, 16), 256, 0, stream>>>(
        A1bf, A2T, Cr, NT, NT, NT, NT, sSb, sSh, sSb, sSh, sSb, sSh, 1.f);
    gemm_async_wmma<4, false, false><<<dim3(6, 6, 16), 256, 0, stream>>>(
        A2bf, A1T, Cl, NT, NT, NT, NT, sSb, sSh, sSb, sSh, sSb, sSh, 1.f);

    // ---- per-edge gate MLP + Smix (in-place over Cr)
    gate_smix_kernel<<<dim3((unsigned)(SZ_S / 256)), 256, 0, stream>>>(
        S1, S2, Cr, Cl, c1w, c1b, c2w, c2b);

    // ---- A = softmax(Smix)
    softmax_rows_kernel<<<dim3(BB * HH * NT), 256, 0, stream>>>(Cr, Abf);

    // ---- transport = A2 @ v2 -> stored transposed (B,H,DK,N) bf16
    gemm_async_wmma<2, true, true><<<dim3(6, 1, 16), 256, 0, stream>>>(
        A2bf, v2T, Tt, NT, NT, NT, NT,
        sSb, sSh, sTb, sTh, sTb, sTh, 1.f);

    // ---- y_chain = A1 @ transport -> (B,N,H,DK) f32
    gemm_async_wmma<2, false, false><<<dim3(6, 1, 16), 256, 0, stream>>>(
        A1bf, Tt, ychain, NT, NT, NT, DM,
        sSb, sSh, sTb, sTh, sYb, sYh, 1.f);

    // ---- y_base = A @ v1 -> (B,N,H,DK) f32
    gemm_async_wmma<2, false, false><<<dim3(6, 1, 16), 256, 0, stream>>>(
        Abf, v1T, ybase, NT, NT, NT, DM,
        sSb, sSh, sTb, sTh, sYb, sYh, 1.f);

    // ---- y = y_base + sigmoid(chain)*y_chain -> bf16
    combine_kernel<<<dim3((unsigned)((SZ_Y + 255) / 256)), 256, 0, stream>>>(
        ybase, ychain, chl, y_bf, SZ_Y);

    // ---- out = y @ proj_w (1536x512x512) -> f32
    gemm_async_wmma<4, false, false><<<dim3(12, 4, 1), 256, 0, stream>>>(
        y_bf, pwT, out, DM, DM, DM, DM, 0, 0, 0, 0, 0, 0, 1.f);

    (void)in_sizes; (void)n_in; (void)out_size; (void)ws_size;
}
